// Int4Linear_30855045054751
// MI455X (gfx1250) — matrix operand order repaired and checked by reading, not compiled, not run on hardware
//
#include <hip/hip_runtime.h>

typedef __attribute__((ext_vector_type(16))) _Float16 v16h;
typedef __attribute__((ext_vector_type(8)))  _Float16 v8h;
typedef __attribute__((ext_vector_type(2)))  _Float16 v2h;
typedef __attribute__((ext_vector_type(8)))  float    v8f;

#define M_DIM 4096          // B*S
#define K_DIM 4096
#define N_DIM 11008
#define BK    64            // k-span per pipeline stage (2 WMMA k-steps)
#define LDT   72            // LDS row stride in halves (144 B, 16B-aligned, padded)
#define KT_STEPS (K_DIM / BK)        // 64
#define GROUPS   (K_DIM / 128)       // 32

#define A_BUF_BYTES (128 * LDT * 2)  // 18432

static __device__ __forceinline__ v16h cat16(v8h lo, v8h hi) {
  return __builtin_shufflevector(lo, hi, 0,1,2,3,4,5,6,7,8,9,10,11,12,13,14,15);
}

// Generic pointers to LDS carry the LDS byte offset in their low 32 bits.
static __device__ __forceinline__ unsigned int lds_off(const void* p) {
  return (unsigned int)(unsigned long long)p;
}

// CDNA5 async DMA: per-lane 16B global -> LDS, tracked with ASYNCcnt.
static __device__ __forceinline__ void async_copy_b128(unsigned int lds_byte_off,
                                                       unsigned long long gaddr) {
  asm volatile("global_load_async_to_lds_b128 %0, %1, off"
               :: "v"(lds_byte_off), "v"(gaddr)
               : "memory");
}

__global__ __launch_bounds__(256)
void int4_linear_wmma_kernel(const _Float16* __restrict__ x,
                             const unsigned char* __restrict__ wp,
                             const _Float16* __restrict__ scales,
                             const _Float16* __restrict__ bias,
                             _Float16* __restrict__ out)
{
  __shared__ __align__(16) _Float16 As[2][128 * LDT];  // 36 KB, DMA double buffer
  __shared__ __align__(16) _Float16 Bs[128 * LDT];     // 18 KB, dequantized weights

  const int tid  = threadIdx.x;
  const int lane = tid & 31;
  const int w    = tid >> 5;     // wave 0..7
  const int wm   = w >> 2;       // 0..1  -> 64-row M slab
  const int wn   = w & 3;        // 0..3  -> 32-col N slab
  const int kh   = lane >> 4;    // lane half
  const int l15  = lane & 15;

  const int mBase = blockIdx.x * 128;
  const int nBase = blockIdx.y * 128;

  // ---- A: async-DMA addressing (1024 x 16B chunks per stage, 4/thread) ----
  unsigned long long aG[4];
  unsigned int       aL[4];
  const unsigned int ldsA = lds_off(&As[0][0]);
  #pragma unroll
  for (int i = 0; i < 4; ++i) {
    int c   = i * 256 + tid;
    int row = c >> 3;            // 0..127
    int col = c & 7;             // 16B chunk within BK
    aG[i] = (unsigned long long)(x + (size_t)(mBase + row) * K_DIM + col * 8);
    aL[i] = ldsA + row * (LDT * 2) + col * 16;
  }

  // ---- W: register-staged packed bytes, block-shared dequant to Bs ----
  const int nRow = tid >> 1;           // 0..127: weight row this thread dequants
  const int hB   = tid & 1;            // which 16B (32 k-values) of the row
  const unsigned char* wGp = wp + (size_t)(nBase + nRow) * (K_DIM / 2) + hB * 16;
  const _Float16*      sGp = scales + (size_t)(nBase + nRow) * GROUPS;
  unsigned int*        bOut = (unsigned int*)&Bs[nRow * LDT + hB * 32];

  v8f acc[4][2];
  #pragma unroll
  for (int i = 0; i < 4; ++i)
    #pragma unroll
    for (int j = 0; j < 2; ++j)
      acc[i][j] = (v8f){0.f,0.f,0.f,0.f,0.f,0.f,0.f,0.f};

  // ---- prologue: DMA A stage 0; stage-0 W bytes + scale into registers ----
  #pragma unroll
  for (int i = 0; i < 4; ++i) async_copy_b128(aL[i], aG[i]);
  uint4    wReg = *(const uint4*)wGp;
  _Float16 sH   = sGp[0];

  int buf = 0;
  for (int kt = 0; kt < KT_STEPS; ++kt) {
    // ---- DMA next A stage into the other buffer, wait for current ----
    if (kt + 1 < KT_STEPS) {
      const unsigned int bofA = (buf ^ 1) * A_BUF_BYTES;
      #pragma unroll
      for (int i = 0; i < 4; ++i)
        async_copy_b128(aL[i] + bofA, aG[i] + (size_t)(kt + 1) * (BK * 2));
      asm volatile("s_wait_asynccnt 4" ::: "memory");   // stage-kt A in LDS
    } else {
      asm volatile("s_wait_asynccnt 0" ::: "memory");
    }
    if (kt + 4 < KT_STEPS) {
      __builtin_prefetch((const void*)(wGp + (size_t)(kt + 4) * (BK / 2)), 0, 1);
    }
    __syncthreads();   // As[buf] visible; all waves done reading Bs (stage kt-1)

    // ---- block-shared dequant: 16 packed bytes -> 32 fp16 into Bs ----
    {
      _Float16 cH = (_Float16)(-1032.0f * (float)sH);  // (q-8)*s = (1024+q)*s - 1032*s
      v2h s2 = (v2h){sH, sH};
      v2h c2 = (v2h){cH, cH};
      unsigned wd[4] = { wReg.x, wReg.y, wReg.z, wReg.w };
      #pragma unroll
      for (int qd = 0; qd < 4; ++qd) {
        unsigned r[4];
        #pragma unroll
        for (int by = 0; by < 4; ++by) {
          unsigned b = (wd[qd] >> (8 * by)) & 0xFFu;
          unsigned t = (((b << 12) | b) & 0x000F000Fu) | 0x64006400u; // {1024+qhi,1024+qlo}
          v2h q = __builtin_bit_cast(v2h, t);
          v2h d = q * s2 + c2;
          r[by] = __builtin_bit_cast(unsigned, d);
        }
        uint4 pk; pk.x = r[0]; pk.y = r[1]; pk.z = r[2]; pk.w = r[3];
        ((uint4*)bOut)[qd] = pk;
      }
    }

    // ---- early global loads for next stage's W bytes + scale ----
    if (kt + 1 < KT_STEPS) {
      wReg = *(const uint4*)(wGp + (size_t)(kt + 1) * (BK / 2));
      sH   = sGp[(kt + 1) >> 1];
    }

    __syncthreads();   // Bs (stage kt) visible to all waves

    // ---- compute: 2 k-steps x (4 M-frags x 2 N-frags) WMMAs ----
    const _Float16* Ab = As[buf];
    #pragma unroll
    for (int ks = 0; ks < BK; ks += 32) {
      v16h bfrag[2];
      #pragma unroll
      for (int j = 0; j < 2; ++j) {
        // B layout: lane holds col n = lane&15, K = ks + 16*kh + e, e=0..15
        const _Float16* p = &Bs[(wn * 32 + j * 16 + l15) * LDT + ks + kh * 16];
        v8h lo = *(const v8h*)p;
        v8h hi = *(const v8h*)(p + 8);
        bfrag[j] = cat16(lo, hi);
      }
      #pragma unroll
      for (int i = 0; i < 4; ++i) {
        // A layout: lane holds row m = lane&15, K = ks + 16*kh + e, e=0..15
        const _Float16* q = &Ab[(wm * 64 + i * 16 + l15) * LDT + ks + kh * 16];
        v8h lo = *(const v8h*)q;
        v8h hi = *(const v8h*)(q + 8);
        v16h afrag = cat16(lo, hi);
        acc[i][0] = __builtin_amdgcn_wmma_f32_16x16x32_f16(
            false, afrag, false, bfrag[0], (short)0, acc[i][0], false, false);
        acc[i][1] = __builtin_amdgcn_wmma_f32_16x16x32_f16(
            false, afrag, false, bfrag[1], (short)0, acc[i][1], false, false);
      }
    }

    buf ^= 1;
  }

  // ---- epilogue: + bias, store fp16 ----
  float bj[2];
  #pragma unroll
  for (int j = 0; j < 2; ++j)
    bj[j] = (float)bias[nBase + wn * 32 + j * 16 + l15];

  #pragma unroll
  for (int i = 0; i < 4; ++i) {
    #pragma unroll
    for (int j = 0; j < 2; ++j) {
      #pragma unroll
      for (int r = 0; r < 8; ++r) {
        int m = mBase + wm * 64 + i * 16 + r + kh * 8;
        int n = nBase + wn * 32 + j * 16 + l15;
        out[(size_t)m * N_DIM + n] = (_Float16)(acc[i][j][r] + bj[j]);
      }
    }
  }
}

extern "C" void kernel_launch(void* const* d_in, const int* in_sizes, int n_in,
                              void* d_out, int out_size, void* d_ws, size_t ws_size,
                              hipStream_t stream) {
  (void)in_sizes; (void)n_in; (void)d_ws; (void)ws_size; (void)out_size;
  const _Float16*      x    = (const _Float16*)d_in[0];
  const unsigned char* wpk  = (const unsigned char*)d_in[1];
  const _Float16*      sc   = (const _Float16*)d_in[2];
  const _Float16*      bias = (const _Float16*)d_in[3];
  _Float16*            out  = (_Float16*)d_out;

  dim3 grid(M_DIM / 128, N_DIM / 128);   // 32 x 86
  int4_linear_wmma_kernel<<<grid, 256, 0, stream>>>(x, wpk, sc, bias, out);
}
